// RoPEAttention_45629732553264
// MI455X (gfx1250) — matrix operand order repaired and checked by reading, not compiled, not run on hardware
//
#include <hip/hip_runtime.h>
#include <hip/hip_bf16.h>

// ---------------------------------------------------------------------------
// RoPE attention (k@q^T, causal, softplus, no normalization) for MI455X.
// B=4, T=4096, C=512, fp32 in/out; internal math in f16 WMMA + f32 accum.
// A-tiles staged to LDS via the Tensor Data Mover (TDM) with LDS padding;
// B-fragment loads software-pipelined against WMMAs with
// __builtin_amdgcn_sched_group_barrier so the matrix pipe stays busy.
// ---------------------------------------------------------------------------

#define BB 4
#define TT 4096
#define CC 512
#define LDP 528   // LDS row stride in halves: 512 + 16 (TDM pad: 8 dwords/row)

typedef _Float16 v8h  __attribute__((ext_vector_type(8)));
typedef _Float16 v16h __attribute__((ext_vector_type(16)));
typedef float    v8f  __attribute__((ext_vector_type(8)));
typedef unsigned int u32x4 __attribute__((ext_vector_type(4)));
typedef int      i32x4 __attribute__((ext_vector_type(4)));
typedef int      i32x8 __attribute__((ext_vector_type(8)));

// --- scheduling directives --------------------------------------------------
#if defined(__has_builtin)
#if __has_builtin(__builtin_amdgcn_sched_group_barrier)
#define SGB(mask, cnt, id) __builtin_amdgcn_sched_group_barrier((mask), (cnt), (id))
#define HAVE_SGB 1
#endif
#endif
#ifndef SGB
#define SGB(mask, cnt, id)
#endif
// masks: 0x008 = MFMA/WMMA, 0x020 = VMEM read, 0x100 = DS read

// Pipeline schedule for a 16-fragment / 16-WMMA GEMM section:
// 4 fragments (8 b128 loads) in flight, then {1 WMMA, 2 loads} x12, drain 4.
__device__ __forceinline__ void sched_gemm16() {
#ifdef HAVE_SGB
  SGB(0x020, 8, 0);
#pragma unroll
  for (int i = 0; i < 12; ++i) {
    SGB(0x008, 1, 0);
    SGB(0x020, 2, 0);
  }
  SGB(0x008, 4, 0);
#endif
}

// --- WMMA fragment loaders (layouts per CDNA5 ISA §7.12.2) ------
// A (16x32 f16, M x K): lanes 0-15 row M=lane, K = {0..7,16..23};
//                       lanes 16-31 row M=lane-16, K = {8..15,24..31}.
__device__ __forceinline__ v16h frag_a(const _Float16* src, int ld, int row0, int k0) {
  const int lane = threadIdx.x & 31;
  const int r = lane & 15, hi = lane >> 4;
  const _Float16* p = src + (size_t)(row0 + r) * ld + k0 + hi * 8;
  v8h lo = *(const v8h*)(p);
  v8h hh = *(const v8h*)(p + 16);
  return __builtin_shufflevector(lo, hh, 0,1,2,3,4,5,6,7,8,9,10,11,12,13,14,15);
}

// B (32x16 f16, K x N): lane n (0-15) holds column N=n, K = {0..7,16..23};
//                       lane n+16 holds column N=n, K = {8..15,24..31}.
// srcT is the (N,K) row-major view (transposed storage), contiguous in K.
__device__ __forceinline__ v16h frag_b(const _Float16* srcT, int ld, int n0, int k0) {
  const int lane = threadIdx.x & 31;
  const int n = lane & 15, hi = lane >> 4;
  const _Float16* p = srcT + (size_t)(n0 + n) * ld + k0 + hi * 8;
  v8h lo = *(const v8h*)(p);
  v8h hh = *(const v8h*)(p + 16);
  return __builtin_shufflevector(lo, hh, 0,1,2,3,4,5,6,7,8,9,10,11,12,13,14,15);
}

__device__ __forceinline__ v8f wmma16(v16h a, v16h b, v8f c) {
  return __builtin_amdgcn_wmma_f32_16x16x32_f16(false, a, false, b, (short)0, c,
                                                false, false);
}

// ---------------------------------------------------------------------------
// Stage a 16x512 f16 tile (16 rows of a (B*T, C) row-major matrix) into LDS
// with row stride LDP=528 halves, using the Tensor Data Mover.
// D# built per CDNA5 ISA:
//   group0: count=1, lds_addr, global_addr, type=2
//   group1: data_size=2B, pad_enable (interval=256 dw, amount=8 dw),
//           tensor_dim0=512, tensor_dim1=B*T, tile=512x16, dim0_stride=512.
// ---------------------------------------------------------------------------
#if defined(__has_builtin)
#if __has_builtin(__builtin_amdgcn_tensor_load_to_lds)
#define HAVE_TDM 1
#endif
#endif

__device__ __forceinline__ void stage_tile16(const _Float16* __restrict__ g,
                                             _Float16* lds) {
#ifdef HAVE_TDM
  if (threadIdx.x < 32) {
    unsigned long long ga = (unsigned long long)(uintptr_t)g;
    unsigned int lds_off = (unsigned int)(uintptr_t)lds;  // LDS aperture: low 32b
    u32x4 g0 = {1u,                                 // count=1 (valid descriptor)
                lds_off,                            // lds_addr
                (unsigned int)ga,                   // global_addr[31:0]
                (((unsigned int)(ga >> 32)) & 0x01FFFFFFu) | 0x80000000u}; // type=2
    i32x8 g1 = {(int)(0x00010000u |                 // data_size = 1 (2 bytes)
                      (1u << 20) |                  // pad_enable
                      (7u << 22) |                  // pad_interval = 256 dwords
                      (7u << 25)),                  // pad_amount   = 8 dwords
                (int)(512u << 16),                  // tensor_dim0[15:0] = 512
                (int)(16384u << 16),                // tensor_dim1[15:0] = B*T
                (int)(512u << 16),                  // tile_dim0 = 512
                16,                                 // tile_dim1 = 16
                512, 0, 0};                         // tensor_dim0_stride = 512
    i32x4 z4 = {0, 0, 0, 0};
#if __clang_major__ >= 23
    i32x8 z8 = {0, 0, 0, 0, 0, 0, 0, 0};
    __builtin_amdgcn_tensor_load_to_lds(g0, g1, z4, z4, z8, 0);
#else
    __builtin_amdgcn_tensor_load_to_lds(g0, g1, z4, z4, 0);
#endif
    __builtin_amdgcn_s_wait_tensorcnt((short)0);
  }
#else
  // Fallback: cooperative block copy (16 rows x 64 chunks of 8 halves).
  for (int i = threadIdx.x; i < 16 * 64; i += 256) {
    int row = i >> 6, col = (i & 63) * 8;
    *(v8h*)&lds[row * LDP + col] = *(const v8h*)&g[(size_t)row * CC + col];
  }
#endif
  __syncthreads();
}

// ---------------------------------------------------------------------------
// Prep kernels: fp32 -> f16 conversion and weight transpose (C,C)->(N,C) f16.
// ---------------------------------------------------------------------------
__global__ void cvt_f16_kernel(const float* __restrict__ src,
                               _Float16* __restrict__ dst, int n) {
  int i = blockIdx.x * blockDim.x + threadIdx.x;
  if (i < n) dst[i] = (_Float16)src[i];
}

__global__ void cvt_wT_kernel(const float* __restrict__ w,
                              _Float16* __restrict__ wt) {
  int i = blockIdx.x * blockDim.x + threadIdx.x;  // i = n*C + c
  if (i < CC * CC) {
    int n = i >> 9;
    int c = i & (CC - 1);
    wt[i] = (_Float16)w[c * CC + n];
  }
}

// ---------------------------------------------------------------------------
// Fused QKV projection + bias + RoPE.  Block = 16 rows of (B*T), 8 waves.
// ---------------------------------------------------------------------------
__global__ __launch_bounds__(256) void qkv_rope_kernel(
    const _Float16* __restrict__ xh,
    const _Float16* __restrict__ wtk, const _Float16* __restrict__ wtq,
    const _Float16* __restrict__ wtv,
    const float* __restrict__ bk, const float* __restrict__ bq,
    const float* __restrict__ bv,
    _Float16* __restrict__ kh, _Float16* __restrict__ qh,
    _Float16* __restrict__ vt) {
  __shared__ __align__(16) _Float16 xs[16 * LDP];

  const int rowg0 = blockIdx.x * 16;       // global row in [0, B*T)
  const int b  = rowg0 / TT;
  const int t0 = rowg0 - b * TT;
  const int wave = threadIdx.x >> 5;
  const int lane = threadIdx.x & 31;
  const int n = lane & 15, hi = lane >> 4;

  stage_tile16(xh + (size_t)rowg0 * CC, xs);

  // Hoist the 16 A-fragments of x from LDS (read once per wave).
  v16h a[16];
#pragma unroll
  for (int kk = 0; kk < 16; ++kk) a[kk] = frag_a(xs, LDP, 0, kk * 32);

#pragma unroll 1
  for (int j = 0; j < 12; ++j) {
    const int tile = wave + 8 * j;   // 0..95
    const int m  = tile >> 5;        // 0:k 1:q 2:v
    const int n0 = (tile & 31) * 16;
    const _Float16* wt = (m == 0) ? wtk : (m == 1) ? wtq : wtv;

    v16h bf[16];
#pragma unroll
    for (int kk = 0; kk < 16; ++kk) bf[kk] = frag_b(wt, CC, n0, kk * 32);
    v8f acc = {};
#pragma unroll
    for (int kk = 0; kk < 16; ++kk) acc = wmma16(a[kk], bf[kk], acc);
    sched_gemm16();   // pipeline: 4 frags in flight ahead of the WMMA stream

    const int ng = n0 + n;
    if (m == 2) {                      // v: bias + transpose store (C,T)
      const float bb = bv[ng];
      _Float16* dst = vt + ((size_t)b * CC + ng) * TT;
#pragma unroll
      for (int i = 0; i < 8; ++i)
        dst[t0 + i + 8 * hi] = (_Float16)(acc[i] + bb);
    } else {                           // k/q: bias + RoPE + row-major store
      const float bb = (m == 0) ? bk[ng] : bq[ng];
      _Float16* dst = (m == 0) ? kh : qh;
      const int d = ng >> 1;
      const bool even = (ng & 1) == 0;
#pragma unroll
      for (int i = 0; i < 8; ++i) {
        const int t = t0 + i + 8 * hi;
        float z  = acc[i] + bb;
        float zo = __shfl_xor(z, 1, 32);           // partner column of pair
        float ang = (float)(TT - 1 - t) * (float)d;
        float sn, cs;
        __sincosf(ang, &sn, &cs);
        float r = even ? (cs * z - sn * zo) : (sn * zo + cs * z);
        dst[(size_t)(rowg0 + i + 8 * hi) * CC + ng] = (_Float16)r;
      }
    }
  }
}

// ---------------------------------------------------------------------------
// Fused causal attention with softplus (streamable: no normalization).
// ---------------------------------------------------------------------------
__global__ __launch_bounds__(256) void attn_kernel(
    const _Float16* __restrict__ kh, const _Float16* __restrict__ qh,
    const _Float16* __restrict__ vt, _Float16* __restrict__ aoh) {
  __shared__ __align__(16) _Float16 ks_lds[16 * LDP];
  __shared__ __align__(16) _Float16 P[16 * 136];  // 16 t rows x 128 s (+pad)

  const int b  = blockIdx.y;
  const int t0 = blockIdx.x * 16;
  const int rowg0 = b * TT + t0;
  const int wave = threadIdx.x >> 5;
  const int lane = threadIdx.x & 31;
  const int n = lane & 15, hi = lane >> 4;
  const float scale = 0.04419417382415922f;  // 512^-0.5

  stage_tile16(kh + (size_t)rowg0 * CC, ks_lds);

  // Hoist K A-fragments for the whole s loop.
  v16h ak[16];
#pragma unroll
  for (int kk = 0; kk < 16; ++kk) ak[kk] = frag_a(ks_lds, LDP, 0, kk * 32);

  v8f o[4] = {v8f{}, v8f{}, v8f{}, v8f{}};
  const int cw = wave * 64;
  const _Float16* vtb = vt + (size_t)b * CC * TT;

  for (int s0 = 0; s0 <= t0 + 15; s0 += 128) {
    // ---- scores: S = K_tile @ Q_s^T for this wave's 16 s columns ----
    const int sq = s0 + wave * 16;
    v16h qb[16];
#pragma unroll
    for (int kk = 0; kk < 16; ++kk)
      qb[kk] = frag_b(qh, CC, b * TT + sq, kk * 32);

    // Speculative prefetch of next step's Q rows (global_prefetch_b8).
    if (s0 + 128 <= t0 + 15)
      __builtin_prefetch(qh + (size_t)(b * TT + sq + 128) * CC, 0, 1);

    v8f sacc = {};
#pragma unroll
    for (int kk = 0; kk < 16; ++kk) sacc = wmma16(ak[kk], qb[kk], sacc);
    sched_gemm16();

    __syncthreads();  // previous iteration's P reads are done

    // ---- mask + softplus -> P in LDS ----
    const int sg = sq + n;
#pragma unroll
    for (int i = 0; i < 8; ++i) {
      const int t = t0 + i + 8 * hi;
      float w = sacc[i] * scale;
      float p = 0.0f;
      if (sg <= t) p = (w > 20.0f) ? w : log1pf(__expf(w));
      P[(i + 8 * hi) * 136 + wave * 16 + n] = (_Float16)p;
    }
    __syncthreads();  // P visible to all waves

    // ---- O += P(16x128) @ V(128x64 chunk) ----
    v16h ap[4];
#pragma unroll
    for (int ksx = 0; ksx < 4; ++ksx) {
      const _Float16* pp = &P[n * 136 + ksx * 32 + hi * 8];
      v8h lo = *(const v8h*)(pp);
      v8h hh = *(const v8h*)(pp + 16);
      ap[ksx] = __builtin_shufflevector(
          lo, hh, 0,1,2,3,4,5,6,7,8,9,10,11,12,13,14,15);
    }
#pragma unroll
    for (int nt = 0; nt < 4; ++nt) {
      v16h bvf[4];
#pragma unroll
      for (int ksx = 0; ksx < 4; ++ksx)
        bvf[ksx] = frag_b(vtb, TT, cw + nt * 16, s0 + ksx * 32);
#pragma unroll
      for (int ksx = 0; ksx < 4; ++ksx) o[nt] = wmma16(ap[ksx], bvf[ksx], o[nt]);
    }
    // PV pipeline: P fragments (8 ds reads) first, then loads ahead of WMMAs.
    SGB(0x100, 8, 0);
    sched_gemm16();
  }

  // ---- store O as f16 (input to final projection) ----
#pragma unroll
  for (int nt = 0; nt < 4; ++nt) {
    const int ng = cw + nt * 16 + n;
#pragma unroll
    for (int i = 0; i < 8; ++i)
      aoh[(size_t)(rowg0 + i + 8 * hi) * CC + ng] = (_Float16)o[nt][i];
  }
}

// ---------------------------------------------------------------------------
// Output projection: out = ao @ Wp + bp, fp32 result.
// ---------------------------------------------------------------------------
__global__ __launch_bounds__(256) void out_proj_kernel(
    const _Float16* __restrict__ aoh, const _Float16* __restrict__ wtp,
    const float* __restrict__ bp, float* __restrict__ out) {
  __shared__ __align__(16) _Float16 as_lds[16 * LDP];

  const int rowg0 = blockIdx.x * 16;
  const int wave = threadIdx.x >> 5;
  const int lane = threadIdx.x & 31;
  const int n = lane & 15, hi = lane >> 4;

  stage_tile16(aoh + (size_t)rowg0 * CC, as_lds);

  v16h a[16];
#pragma unroll
  for (int kk = 0; kk < 16; ++kk) a[kk] = frag_a(as_lds, LDP, 0, kk * 32);

  const int n0 = wave * 64;
#pragma unroll 1
  for (int nt = 0; nt < 4; ++nt) {
    v16h bf[16];
#pragma unroll
    for (int kk = 0; kk < 16; ++kk)
      bf[kk] = frag_b(wtp, CC, n0 + nt * 16, kk * 32);
    v8f acc = {};
#pragma unroll
    for (int kk = 0; kk < 16; ++kk) acc = wmma16(a[kk], bf[kk], acc);
    sched_gemm16();

    const int ng = n0 + nt * 16 + n;
    const float bb = bp[ng];
#pragma unroll
    for (int i = 0; i < 8; ++i)
      out[(size_t)(rowg0 + i + 8 * hi) * CC + ng] = acc[i] + bb;
  }
}

// ---------------------------------------------------------------------------
// Host-side launcher.
// ---------------------------------------------------------------------------
extern "C" void kernel_launch(void* const* d_in, const int* in_sizes, int n_in,
                              void* d_out, int out_size, void* d_ws,
                              size_t ws_size, hipStream_t stream) {
  (void)in_sizes; (void)n_in; (void)out_size; (void)ws_size;

  const float* x  = (const float*)d_in[0];
  const float* Wk = (const float*)d_in[1];
  const float* bk = (const float*)d_in[2];
  const float* Wq = (const float*)d_in[3];
  const float* bq = (const float*)d_in[4];
  const float* Wv = (const float*)d_in[5];
  const float* bv = (const float*)d_in[6];
  const float* Wp = (const float*)d_in[7];
  const float* bp = (const float*)d_in[8];
  float* out = (float*)d_out;

  const size_t BTC = (size_t)BB * TT * CC;   // 8,388,608
  size_t off = 0;
  auto alloc = [&](size_t bytes) -> void* {
    void* p = (char*)d_ws + off;
    off += (bytes + 255) & ~(size_t)255;
    return p;
  };
  _Float16* xh  = (_Float16*)alloc(BTC * 2);
  _Float16* wtk = (_Float16*)alloc((size_t)CC * CC * 2);
  _Float16* wtq = (_Float16*)alloc((size_t)CC * CC * 2);
  _Float16* wtv = (_Float16*)alloc((size_t)CC * CC * 2);
  _Float16* wtp = (_Float16*)alloc((size_t)CC * CC * 2);
  _Float16* kh  = (_Float16*)alloc(BTC * 2);
  _Float16* qh  = (_Float16*)alloc(BTC * 2);
  _Float16* vt  = (_Float16*)alloc(BTC * 2);   // transposed (B, C, T)
  _Float16* aoh = (_Float16*)alloc(BTC * 2);

  // 1) fp32 -> f16 conversions / weight transposes.
  cvt_f16_kernel<<<(int)((BTC + 255) / 256), 256, 0, stream>>>(x, xh, (int)BTC);
  const int wblk = (CC * CC + 255) / 256;
  cvt_wT_kernel<<<wblk, 256, 0, stream>>>(Wk, wtk);
  cvt_wT_kernel<<<wblk, 256, 0, stream>>>(Wq, wtq);
  cvt_wT_kernel<<<wblk, 256, 0, stream>>>(Wv, wtv);
  cvt_wT_kernel<<<wblk, 256, 0, stream>>>(Wp, wtp);

  // 2) fused QKV + bias + RoPE.
  qkv_rope_kernel<<<BB * TT / 16, 256, 0, stream>>>(
      xh, wtk, wtq, wtv, bk, bq, bv, kh, qh, vt);

  // 3) fused causal softplus attention.
  attn_kernel<<<dim3(TT / 16, BB), 256, 0, stream>>>(kh, qh, vt, aoh);

  // 4) output projection.
  out_proj_kernel<<<BB * TT / 16, 256, 0, stream>>>(aoh, wtp, bp, out);
}
